// MultiHeadAttention_19258633355608
// MI455X (gfx1250) — matrix operand order repaired and checked by reading, not compiled, not run on hardware
//
#include <hip/hip_runtime.h>

// ---------------- problem constants ----------------
#define DM   2048     // d_model
#define NH   16       // heads
#define DH   128      // head dim
#define BB   2        // batch
#define SS   2048     // seq
#define MROWS (BB*SS) // 4096

typedef __bf16  bfx8  __attribute__((ext_vector_type(8)));
typedef __bf16  v16bf __attribute__((ext_vector_type(16)));
typedef float   v8f   __attribute__((ext_vector_type(8)));
typedef float          f32x4 __attribute__((ext_vector_type(4)));
typedef unsigned short u16x4 __attribute__((ext_vector_type(4)));

// explicit global-address-space views: guarantees global_load/global_store
// (no flat path, no DScnt coupling, no lost addrspace inference)
typedef __attribute__((address_space(1))) const unsigned short g_cu16;
typedef __attribute__((address_space(1)))       unsigned short g_u16;
typedef __attribute__((address_space(1))) const float          g_cf32;
typedef __attribute__((address_space(1)))       float          g_f32;
typedef __attribute__((address_space(1))) const bfx8           g_bfx8;
typedef __attribute__((address_space(1))) const v16bf          g_v16bf;
typedef __attribute__((address_space(1))) const f32x4          g_f32x4;
typedef __attribute__((address_space(1)))       u16x4          g_u16x4;

#define GCP(p) ((g_cu16*)(unsigned long long)(p))
#define GMP(p) ((g_u16*)(unsigned long long)(p))

union AV16 { v16bf v; bfx8 h[2]; };

__device__ __forceinline__ unsigned short f2bf(float f) {
    union { float f; unsigned u; } x; x.f = f;
    unsigned r = x.u + 0x7FFFu + ((x.u >> 16) & 1u);   // round-to-nearest-even
    return (unsigned short)(r >> 16);
}

__device__ __forceinline__ float hmax16(float v) {
    #pragma unroll
    for (int o = 1; o < 16; o <<= 1) v = fmaxf(v, __shfl_xor(v, o, 16));
    return v;
}
__device__ __forceinline__ float hsum16(float v) {
    #pragma unroll
    for (int o = 1; o < 16; o <<= 1) v += __shfl_xor(v, o, 16);
    return v;
}

// ---------------- stage 1: fp32 -> bf16 convert ----------------
__global__ void cvt_f32_bf16(const float* __restrict__ in,
                             unsigned short* __restrict__ out, int n4) {
    int i = blockIdx.x * blockDim.x + threadIdx.x;
    if (i >= n4) return;
    const f32x4 f = ((g_f32x4*)(unsigned long long)in)[i];
    u16x4 o;
    o.x = f2bf(f.x); o.y = f2bf(f.y); o.z = f2bf(f.z); o.w = f2bf(f.w);
    ((g_u16x4*)(unsigned long long)out)[i] = o;
}

// transpose 2048x2048 fp32 -> bf16 [N][K]
__global__ void transpose_cvt(const float* __restrict__ W,
                              unsigned short* __restrict__ Wt) {
    int idx = blockIdx.x * blockDim.x + threadIdx.x;   // over 2048*2048
    int k = idx >> 11;          // row of W
    int n = idx & 2047;         // col of W
    GMP(Wt)[(size_t)n * DM + k] = f2bf(((g_cf32*)(unsigned long long)W)[idx]);
}

// ---------------- stage 2/4: GEMM  C[M,N] = A[M,K] * Wt[N,K]^T + bias ------
// wave-tile: 32 rows x 64 cols (2x4 WMMA grid), K unrolled by 2 (64 K/iter):
// 24 global loads clause at the top of the body, 16 WMMAs consume them with
// progressive loadcnt waits. Scalar base pointers + immediate offsets only.
// mode 0: bf16 Q [b,h,s,d] | 1: bf16 K [b,h,t,d] | 2: bf16 Vt [b,h,d,t]
// mode 3: fp32 row-major (final output)
__global__ void __launch_bounds__(256)
gemm_bf16(const unsigned short* __restrict__ A,
          const unsigned short* __restrict__ Wt,
          const float* __restrict__ bias,
          void* __restrict__ out, int mode) {
    const int lane = threadIdx.x & 31;
    const int m    = lane & 15;
    const int hi   = lane >> 4;
    const int wave = (blockIdx.x * (blockDim.x >> 5)) + (threadIdx.x >> 5);

    const int tilesN = DM / 64;                  // 32
    const int m0 = (wave / tilesN) * 32;
    const int n0 = (wave % tilesN) * 64;

    v8f acc[2][4];
    #pragma unroll
    for (int i = 0; i < 2; ++i)
        #pragma unroll
        for (int j = 0; j < 4; ++j) acc[i][j] = (v8f)(0.0f);

    g_cu16* a0 = GCP(A)  + (size_t)(m0 + m) * DM + 16 * hi;
    g_cu16* a1 = a0 + (size_t)16 * DM;
    g_cu16* b0 = GCP(Wt) + (size_t)(n0 + m) * DM + 16 * hi;
    g_cu16* b1 = b0 + (size_t)16 * DM;
    g_cu16* b2 = b0 + (size_t)32 * DM;
    g_cu16* b3 = b0 + (size_t)48 * DM;

    #pragma unroll 1
    for (int kk = 0; kk < DM / 64; ++kk) {       // 32 iters, 64 K each
        AV16 aE[2], aO[2];
        v16bf bE[4], bO[4];
        // even half (K offset 0..31): A chunks at +0/+8, B at +0
        aE[0].h[0] = *(g_bfx8*)(a0);      aE[0].h[1] = *(g_bfx8*)(a0 + 8);
        aE[1].h[0] = *(g_bfx8*)(a1);      aE[1].h[1] = *(g_bfx8*)(a1 + 8);
        bE[0] = *(g_v16bf*)(b0);          bE[1] = *(g_v16bf*)(b1);
        bE[2] = *(g_v16bf*)(b2);          bE[3] = *(g_v16bf*)(b3);
        // odd half (K offset 32..63): +32/+40, B at +32
        aO[0].h[0] = *(g_bfx8*)(a0 + 32); aO[0].h[1] = *(g_bfx8*)(a0 + 40);
        aO[1].h[0] = *(g_bfx8*)(a1 + 32); aO[1].h[1] = *(g_bfx8*)(a1 + 40);
        bO[0] = *(g_v16bf*)(b0 + 32);     bO[1] = *(g_v16bf*)(b1 + 32);
        bO[2] = *(g_v16bf*)(b2 + 32);     bO[3] = *(g_v16bf*)(b3 + 32);

        #pragma unroll
        for (int i = 0; i < 2; ++i)
            #pragma unroll
            for (int j = 0; j < 4; ++j)
                acc[i][j] = __builtin_amdgcn_wmma_f32_16x16x32_bf16(
                    false, aE[i].v, false, bE[j], (short)0, acc[i][j], false, false);
        #pragma unroll
        for (int i = 0; i < 2; ++i)
            #pragma unroll
            for (int j = 0; j < 4; ++j)
                acc[i][j] = __builtin_amdgcn_wmma_f32_16x16x32_bf16(
                    false, aO[i].v, false, bO[j], (short)0, acc[i][j], false, false);

        a0 += 64; a1 += 64; b0 += 64; b1 += 64; b2 += 64; b3 += 64;
    }

    // epilogue: bias + scatter store
    #pragma unroll
    for (int i = 0; i < 2; ++i) {
        #pragma unroll
        for (int j = 0; j < 4; ++j) {
            const int gc = n0 + j * 16 + m;
            const float bv = ((g_cf32*)(unsigned long long)bias)[gc];
            #pragma unroll
            for (int r = 0; r < 8; ++r) {
                const int gr = m0 + i * 16 + r + 8 * hi;
                const float val = acc[i][j][r] + bv;
                const int b = gr >> 11, s = gr & (SS - 1);
                const int h = gc >> 7,  d = gc & (DH - 1);
                if (mode == 3) {
                    ((g_f32*)(unsigned long long)out)[(size_t)gr * DM + gc] = val;
                } else if (mode == 2) {
                    GMP(out)[(((size_t)b * NH + h) * DH + d) * SS + s] = f2bf(val);
                } else {
                    GMP(out)[(((size_t)b * NH + h) * SS + s) * DH + d] = f2bf(val);
                }
            }
        }
    }
}

// ---------------- stage 3: causal flash attention ----------------
// one wave = one 16-row query block of one (b,h); key blocks of 32.
__global__ void __launch_bounds__(128)
flash_attn(const unsigned short* __restrict__ Q,
           const unsigned short* __restrict__ K,
           const unsigned short* __restrict__ Vt,
           unsigned short* __restrict__ ctx) {
    __shared__ __align__(16) unsigned short pbuf[4][16 * 40];

    const int lane = threadIdx.x & 31;
    const int m    = lane & 15;
    const int hi   = lane >> 4;
    const int wv   = threadIdx.x >> 5;
    const int tile = blockIdx.x * 4 + wv;

    const int bh = tile >> 7;            // S/16 = 128 query blocks per (b,h)
    const int s0 = (tile & 127) * 16;
    const int b  = bh >> 4, h = bh & 15;

    const float sm_scale = 0.08838834764831845f;   // 1/sqrt(128)

    // Q A-tiles resident in registers: 4 K-steps of 32
    g_cu16* qrow = GCP(Q) + ((size_t)bh * SS + s0 + m) * DH;
    AV16 qa[4];
    #pragma unroll
    for (int kk = 0; kk < 4; ++kk) {
        qa[kk].h[0] = *(g_bfx8*)(qrow + kk * 32 + 16 * hi);
        qa[kk].h[1] = *(g_bfx8*)(qrow + kk * 32 + 8 + 16 * hi);
    }

    v8f cacc[8];
    #pragma unroll
    for (int n = 0; n < 8; ++n) cacc[n] = (v8f)(0.0f);
    float mrow[8], lrow[8];
    #pragma unroll
    for (int r = 0; r < 8; ++r) { mrow[r] = -__builtin_inff(); lrow[r] = 0.0f; }

    g_cu16* kbase  = GCP(K)  + ((size_t)bh * SS + m) * DH + 16 * hi;
    g_cu16* vtbase = GCP(Vt) + ((size_t)bh * DH + m) * SS + 16 * hi;

    const int nblk = ((s0 + 15) >> 5) + 1;
    #pragma unroll 1
    for (int kb = 0; kb < nblk; ++kb) {
        const int t0 = kb * 32;

        // ---- batched K-tile loads (one clause), then 8 score WMMAs ----
        v16bf kb0[4], kb1[4];
        {
            g_cu16* kr0 = kbase + (size_t)t0 * DH;
            g_cu16* kr1 = kr0 + 16 * DH;
            #pragma unroll
            for (int kk = 0; kk < 4; ++kk) {
                kb0[kk] = *(g_v16bf*)(kr0 + kk * 32);
                kb1[kk] = *(g_v16bf*)(kr1 + kk * 32);
            }
        }
        v8f sA = (v8f)(0.0f), sB = (v8f)(0.0f);
        #pragma unroll
        for (int kk = 0; kk < 4; ++kk) {
            sA = __builtin_amdgcn_wmma_f32_16x16x32_bf16(false, qa[kk].v, false, kb0[kk],
                                                         (short)0, sA, false, false);
            sB = __builtin_amdgcn_wmma_f32_16x16x32_bf16(false, qa[kk].v, false, kb1[kk],
                                                         (short)0, sB, false, false);
        }

        // ---- batched V-tile loads (overlap with softmax VALU below) ----
        v16bf vb[8];
        #pragma unroll
        for (int n = 0; n < 8; ++n)
            vb[n] = *(g_v16bf*)(vtbase + (size_t)(n * 16) * SS + t0);

        // ---- online softmax over the 32 columns of this block ----
        const int  srow_base = s0 + 8 * hi;
        const bool need_mask = (t0 + 31 > s0);
        float p0[8], p1[8];
        #pragma unroll
        for (int r = 0; r < 8; ++r) {
            const int srow = srow_base + r;
            float v0 = sA[r] * sm_scale;
            float v1 = sB[r] * sm_scale;
            if (need_mask) {
                if (t0 + m      > srow) v0 = -__builtin_inff();
                if (t0 + 16 + m > srow) v1 = -__builtin_inff();
            }
            float bm = hmax16(fmaxf(v0, v1));
            const float mnew = fmaxf(mrow[r], bm);
            const float rescale = __expf(mrow[r] - mnew);
            mrow[r] = mnew;
            p0[r] = __expf(v0 - mnew);
            p1[r] = __expf(v1 - mnew);
            const float bs = hsum16(p0[r] + p1[r]);
            lrow[r] = lrow[r] * rescale + bs;
            #pragma unroll
            for (int n = 0; n < 8; ++n) cacc[n][r] *= rescale;
        }

        // ---- P -> LDS (C-layout) then reload in A-layout ----
        unsigned short* pw = &pbuf[wv][0];
        #pragma unroll
        for (int r = 0; r < 8; ++r) {
            pw[(r + 8 * hi) * 40 + m]      = f2bf(p0[r]);
            pw[(r + 8 * hi) * 40 + 16 + m] = f2bf(p1[r]);
        }
        AV16 pa;  // A-tile 16x32: lane m = row; K contiguous 16 per half
        pa.h[0] = *(const bfx8*)(pw + m * 40 + 16 * hi);
        pa.h[1] = *(const bfx8*)(pw + m * 40 + 8 + 16 * hi);

        // ---- context += P * V  (V stored transposed [b,h,d,t]) ----
        #pragma unroll
        for (int n = 0; n < 8; ++n)
            cacc[n] = __builtin_amdgcn_wmma_f32_16x16x32_bf16(
                          false, pa.v, false, vb[n], (short)0, cacc[n], false, false);
    }

    // ---- normalize and store ctx as bf16 [b, s, h*DH] ----
    float inv[8];
    #pragma unroll
    for (int r = 0; r < 8; ++r) inv[r] = 1.0f / lrow[r];
    #pragma unroll
    for (int n = 0; n < 8; ++n) {
        #pragma unroll
        for (int r = 0; r < 8; ++r) {
            const int srow = s0 + r + 8 * hi;
            GMP(ctx)[((size_t)b * SS + srow) * DM + h * DH + n * 16 + m] =
                f2bf(cacc[n][r] * inv[r]);
        }
    }
}

// ---------------- host-side orchestration ----------------
extern "C" void kernel_launch(void* const* d_in, const int* in_sizes, int n_in,
                              void* d_out, int out_size, void* d_ws, size_t ws_size,
                              hipStream_t stream) {
    const float* x  = (const float*)d_in[0];
    const float* Wq = (const float*)d_in[1];
    const float* bq = (const float*)d_in[2];
    const float* Wk = (const float*)d_in[3];
    const float* bk = (const float*)d_in[4];
    const float* Wv = (const float*)d_in[5];
    const float* bv = (const float*)d_in[6];
    const float* Wo = (const float*)d_in[7];
    const float* bo = (const float*)d_in[8];

    // workspace layout (bf16 element counts)
    unsigned short* ws  = (unsigned short*)d_ws;
    const size_t NXB = (size_t)MROWS * DM;   // 8M elems
    const size_t NW  = (size_t)DM * DM;      // 4M elems
    unsigned short* xb  = ws;
    unsigned short* wqt = xb  + NXB;
    unsigned short* wkt = wqt + NW;
    unsigned short* wvt = wkt + NW;
    unsigned short* wot = wvt + NW;
    unsigned short* qb  = wot + NW;
    unsigned short* kb  = qb  + NXB;
    unsigned short* vtb = kb  + NXB;
    unsigned short* ctx = vtb + NXB;

    // 1) convert / transpose to bf16
    cvt_f32_bf16<<<(int)(NXB / 4 / 256), 256, 0, stream>>>(x, xb, (int)(NXB / 4));
    const int tgrid = (int)(NW / 256);
    transpose_cvt<<<tgrid, 256, 0, stream>>>(Wq, wqt);
    transpose_cvt<<<tgrid, 256, 0, stream>>>(Wk, wkt);
    transpose_cvt<<<tgrid, 256, 0, stream>>>(Wv, wvt);
    transpose_cvt<<<tgrid, 256, 0, stream>>>(Wo, wot);

    // 2) projections: (4096/32)*(2048/64) = 4096 wave-tiles, 8 waves/block
    const int gblocks = (MROWS / 32) * (DM / 64) / 8;   // 512
    gemm_bf16<<<gblocks, 256, 0, stream>>>(xb, wqt, bq, qb,  0);
    gemm_bf16<<<gblocks, 256, 0, stream>>>(xb, wkt, bk, kb,  1);
    gemm_bf16<<<gblocks, 256, 0, stream>>>(xb, wvt, bv, vtb, 2);

    // 3) causal flash attention: B*H*(S/16) = 4096 wave-tiles, 4 waves/block
    flash_attn<<<(BB * NH * (SS / 16)) / 4, 128, 0, stream>>>(qb, kb, vtb, ctx);

    // 4) output projection -> fp32 d_out
    gemm_bf16<<<gblocks, 256, 0, stream>>>(ctx, wot, bo, d_out, 3);
}
